// JointModel_19464791785965
// MI455X (gfx1250) — matrix operand order repaired and checked by reading, not compiled, not run on hardware
//
#include <hip/hip_runtime.h>
#include <hip/hip_bf16.h>
#include <stdint.h>

// Problem constants (from reference)
#define P_TOK 4096
#define T_TOK 1024
#define DIM   2048
#define VOC   32000
#define HEADS 8
#define DHEAD 256
#define FFDIM 8192
#define NLAY  2

typedef __attribute__((ext_vector_type(16))) __bf16 v16bf;
typedef __attribute__((ext_vector_type(8)))  float  v8f;

union ABf {
  v16bf v;
  uint32_t u[8];
  unsigned short s[16];
};

union U4S {
  uint4 q;
  unsigned short s[8];
};

// Native conversions (backend picks hardware cvt; avoids manual RNE VALU chains)
__device__ __forceinline__ unsigned short f2bf(float f) {
  union { __bf16 h; unsigned short s; } u;
  u.h = (__bf16)f;
  return u.s;
}
__device__ __forceinline__ float bf2f(unsigned short s) {
  return __uint_as_float(((uint32_t)s) << 16);
}

// gfx1250 async global->LDS copy (ASYNCcnt-tracked Tensor/async path).
// VDST holds the LDS byte address, VADDR the 64-bit global address.
__device__ __forceinline__ void async_ld_b128(const unsigned short* gsrc,
                                              unsigned short* ldst) {
  unsigned loff = (unsigned)(size_t)ldst;  // flat->LDS: hw truncates to LDS offset
  asm volatile("global_load_async_to_lds_b128 %0, %1, off"
               :: "v"(loff), "v"(gsrc) : "memory");
}
__device__ __forceinline__ void async_wait0() {
  asm volatile("s_wait_asynccnt 0x0" ::: "memory");
}

// ---------------------------------------------------------------------------
// Tiled bf16 WMMA GEMM: C[M,N] = A[M,K]_bf16 @ B_bf16, C fp32 or bf16.
// btrans=0: B stored [K,N]; btrans=1: B stored [N,K] (C = A @ B^T).
// Block 128x128x32, 256 threads = 8 wave32 waves (4x2), wave tile 32x64.
// A (and B when btrans) tiles staged with global_load_async_to_lds_b128.
// ---------------------------------------------------------------------------
#define GBM 128
#define GBN 128
#define GBK 32
#define GPAD 8

__global__ __launch_bounds__(256) void k_gemm_bf16(
    const unsigned short* __restrict__ A, const unsigned short* __restrict__ B,
    void* __restrict__ Cv, int M, int N, int K, int btrans, int cbf16) {
  __shared__ __align__(16) unsigned short Al[GBM][GBK + GPAD];
  __shared__ __align__(16) unsigned short Bl[GBN][GBK + GPAD];
  const int tid  = threadIdx.x;
  const int lane = tid & 31;
  const int wid  = tid >> 5;
  const int wm   = wid >> 1;       // 0..3 (M)
  const int wn   = wid & 1;        // 0..1 (N)
  const int mlo  = lane & 15;
  const int hi   = lane >> 4;
  const int rowBase = blockIdx.y * GBM;
  const int colBase = blockIdx.x * GBN;

  v8f acc[2][4];
  for (int i = 0; i < 2; ++i)
    for (int j = 0; j < 4; ++j)
      for (int r = 0; r < 8; ++r) acc[i][j][r] = 0.0f;

  const int srow = tid >> 1;          // 0..127
  const int shalf = (tid & 1) * 16;   // 0/16

  for (int k0 = 0; k0 < K; k0 += GBK) {
    if (k0 + GBK < K)  // hint next A tile into cache (global_prefetch_b8)
      __builtin_prefetch(A + (size_t)(rowBase + srow) * K + k0 + GBK, 0, 0);
    // stage A tile (async copy, bf16 bytes, 32B per thread)
    {
      const unsigned short* src = A + (size_t)(rowBase + srow) * K + k0 + shalf;
      async_ld_b128(src, &Al[srow][shalf]);
      async_ld_b128(src + 8, &Al[srow][shalf + 8]);
    }
    if (btrans) {
      const unsigned short* src = B + (size_t)(colBase + srow) * K + k0 + shalf;
      async_ld_b128(src, &Bl[srow][shalf]);
      async_ld_b128(src + 8, &Bl[srow][shalf + 8]);
    } else {
      // B[K,N] -> Bl[n][k] transpose (strided; element-wise)
      int n  = tid >> 1;
      int ks = (tid & 1) * 16;
      const unsigned short* src = B + (size_t)(k0 + ks) * N + colBase + n;
      for (int kk = 0; kk < 16; ++kk)
        Bl[n][ks + kk] = src[(size_t)kk * N];
    }
    async_wait0();
    __syncthreads();

    // fragments (A: hi half -> K {8h..8h+7}+{16+8h..}; B: hi half -> K {8h..8h+7}+{16+8h..})
    ABf af[2], bf[4];
    for (int i = 0; i < 2; ++i) {
      const unsigned short* p = &Al[wm * 32 + i * 16 + mlo][hi * 8];
      *(uint4*)&af[i].u[0] = *(const uint4*)p;
      *(uint4*)&af[i].u[4] = *(const uint4*)(p + 16);
    }
    for (int j = 0; j < 4; ++j) {
      const unsigned short* p = &Bl[wn * 64 + j * 16 + mlo][hi * 8];
      *(uint4*)&bf[j].u[0] = *(const uint4*)p;
      *(uint4*)&bf[j].u[4] = *(const uint4*)(p + 16);
    }
    for (int i = 0; i < 2; ++i)
      for (int j = 0; j < 4; ++j)
        acc[i][j] = __builtin_amdgcn_wmma_f32_16x16x32_bf16(
            false, af[i].v, false, bf[j].v, (short)0, acc[i][j], false, false);
    __syncthreads();
  }

  if (cbf16) {
    unsigned short* C = (unsigned short*)Cv;
    for (int i = 0; i < 2; ++i)
      for (int j = 0; j < 4; ++j) {
        int col = colBase + wn * 64 + j * 16 + mlo;
        for (int r = 0; r < 8; ++r) {
          int row = rowBase + wm * 32 + i * 16 + hi * 8 + r;
          C[(size_t)row * N + col] = f2bf(acc[i][j][r]);
        }
      }
  } else {
    float* C = (float*)Cv;
    for (int i = 0; i < 2; ++i)
      for (int j = 0; j < 4; ++j) {
        int col = colBase + wn * 64 + j * 16 + mlo;
        for (int r = 0; r < 8; ++r) {
          int row = rowBase + wm * 32 + i * 16 + hi * 8 + r;
          C[(size_t)row * N + col] = acc[i][j][r];
        }
      }
  }
}

// ---------------------------------------------------------------------------
// Flash attention over bf16 QKV (one head per blockIdx.y, 64 q rows per block,
// 4 waves x 16 q rows). DH = 256. Online softmax; masks computed on the fly.
// mode 0: prefix-causal. mode 1: draft block-sparse (anchor / 16-block rules).
// Output bf16 (consumed by the next WMMA GEMM).
// ---------------------------------------------------------------------------
#define FQ 64
#define FKV 64
#define FPAD 8

__global__ __launch_bounds__(128) void k_flash(
    const unsigned short* __restrict__ Qb, int ldq,
    const unsigned short* __restrict__ Kb, int ldk,
    const unsigned short* __restrict__ Vb, int ldv,
    unsigned short* __restrict__ Ob, int ldo,
    int nq, int nkv,
    const int* __restrict__ q_bid, const int* __restrict__ q_pos,
    const int* __restrict__ k_bid, const int* __restrict__ k_pos,
    int mode, int prefLen, float scale) {
  (void)nq;
  __shared__ __align__(16) unsigned short Vt[DHEAD][FKV + FPAD];  // V^T tile
  __shared__ __align__(16) unsigned short Pl[FQ][FKV + FPAD];     // P tile (bf16)
  const int tid  = threadIdx.x;
  const int lane = tid & 31;
  const int wid  = tid >> 5;
  const int mlo  = lane & 15;
  const int hi   = lane >> 4;
  const int hoff  = blockIdx.y * DHEAD;
  const int qrow0 = blockIdx.x * FQ + wid * 16;

  // Preload Q fragments (A layout) directly from bf16 global
  ABf qf[8];
  {
    const unsigned short* qr = Qb + (size_t)(qrow0 + mlo) * ldq + hoff;
    for (int kk = 0; kk < 8; ++kk) {
      const unsigned short* p = qr + kk * 32 + hi * 8;
      *(uint4*)&qf[kk].u[0] = *(const uint4*)p;
      *(uint4*)&qf[kk].u[4] = *(const uint4*)(p + 16);
    }
  }
  // Per-lane metadata for the 8 C-layout rows this lane owns (row = qrow0+8*hi+r)
  int rqb[8], rqp[8], ranch[8], rblk[8];
  for (int r = 0; r < 8; ++r) {
    int q = qrow0 + 8 * hi + r;
    rqb[r]   = q_bid[q];
    rqp[r]   = q_pos[q];
    rblk[r]  = q >> 4;
    ranch[r] = q_pos[(q >> 4) << 4];
  }
  float mrow[8], lrow[8];
  for (int r = 0; r < 8; ++r) { mrow[r] = -3.0e38f; lrow[r] = 0.0f; }
  v8f o[16];
  for (int nt = 0; nt < 16; ++nt)
    for (int r = 0; r < 8; ++r) o[nt][r] = 0.0f;

  for (int kv0 = 0; kv0 < nkv; kv0 += FKV) {
    __syncthreads();
    // Cooperative transposed stage of V tile: Vt[dh][kv]
    for (int it = 0; it < 16; ++it) {
      int flat = tid * 16 + it;      // 2048 uint4 slots = 64 rows * 32 groups
      int krow = flat >> 5;          // 0..63
      int cg   = (flat & 31) * 8;    // dh column group of 8
      U4S w;
      w.q = *(const uint4*)(Vb + (size_t)(kv0 + krow) * ldv + hoff + cg);
      for (int b = 0; b < 8; ++b) Vt[cg + b][krow] = w.s[b];
    }
    __syncthreads();

    // S = Q @ K^T (16x64 per wave); K rows contraction-contiguous -> direct B frags
    float sv[4][8];
    for (int j = 0; j < 4; ++j) {
      v8f s;
      for (int r = 0; r < 8; ++r) s[r] = 0.0f;
      const unsigned short* kr = Kb + (size_t)(kv0 + j * 16 + mlo) * ldk + hoff;
      for (int kk = 0; kk < 8; ++kk) {
        ABf bf;
        const unsigned short* p = kr + kk * 32 + hi * 8;
        *(uint4*)&bf.u[0] = *(const uint4*)p;
        *(uint4*)&bf.u[4] = *(const uint4*)(p + 16);
        s = __builtin_amdgcn_wmma_f32_16x16x32_bf16(
            false, qf[kk].v, false, bf.v, (short)0, s, false, false);
      }
      for (int r = 0; r < 8; ++r) sv[j][r] = s[r] * scale;
    }
    // Mask
    for (int j = 0; j < 4; ++j) {
      int kidx = kv0 + j * 16 + mlo;
      int kb = k_bid[kidx];
      int kp = k_pos[kidx];
      for (int r = 0; r < 8; ++r) {
        bool ok;
        if (mode == 0) {
          ok = (rqb[r] == kb) && (rqp[r] >= kp);
        } else {
          bool pv = (kidx < prefLen) && (ranch[r] > kp);
          bool tb = (kidx >= prefLen) && (rblk[r] == ((kidx - prefLen) >> 4));
          ok = (rqb[r] == kb) && (pv || tb);
        }
        if (!ok) sv[j][r] = -1.0e30f;
      }
    }
    // Online softmax update
    float mnew[8], alpha[8], rs[8];
    for (int r = 0; r < 8; ++r) {
      float v = sv[0][r];
      for (int j = 1; j < 4; ++j) v = fmaxf(v, sv[j][r]);
      for (int d = 1; d < 16; d <<= 1) v = fmaxf(v, __shfl_xor(v, d, 32));
      mnew[r]  = fmaxf(mrow[r], v);
      alpha[r] = __expf(mrow[r] - mnew[r]);
      rs[r] = 0.0f;
    }
    for (int j = 0; j < 4; ++j)
      for (int r = 0; r < 8; ++r) {
        float p = __expf(sv[j][r] - mnew[r]);
        rs[r] += p;
        Pl[wid * 16 + 8 * hi + r][j * 16 + mlo] = f2bf(p);
      }
    for (int r = 0; r < 8; ++r) {
      float v = rs[r];
      for (int d = 1; d < 16; d <<= 1) v += __shfl_xor(v, d, 32);
      lrow[r] = lrow[r] * alpha[r] + v;
      mrow[r] = mnew[r];
    }
    for (int nt = 0; nt < 16; ++nt)
      for (int r = 0; r < 8; ++r) o[nt][r] *= alpha[r];
    __syncthreads();  // publish P (C-layout -> A-layout via LDS)
    // O += P @ V
    for (int kk = 0; kk < 2; ++kk) {
      ABf pa;
      const unsigned short* pr = &Pl[wid * 16 + mlo][kk * 32 + hi * 8];
      *(uint4*)&pa.u[0] = *(const uint4*)pr;
      *(uint4*)&pa.u[4] = *(const uint4*)(pr + 16);
      for (int nt = 0; nt < 16; ++nt) {
        ABf vb;
        const unsigned short* vp = &Vt[nt * 16 + mlo][kk * 32 + hi * 8];
        *(uint4*)&vb.u[0] = *(const uint4*)vp;
        *(uint4*)&vb.u[4] = *(const uint4*)(vp + 16);
        o[nt] = __builtin_amdgcn_wmma_f32_16x16x32_bf16(
            false, pa.v, false, vb.v, (short)0, o[nt], false, false);
      }
    }
  }
  // epilogue: normalize and store bf16
  for (int nt = 0; nt < 16; ++nt) {
    int col = hoff + nt * 16 + mlo;
    for (int r = 0; r < 8; ++r) {
      int row = qrow0 + 8 * hi + r;
      Ob[(size_t)row * ldo + col] = f2bf(o[nt][r] / lrow[r]);
    }
  }
}

// ---------------------------------------------------------------------------
// Elementwise / reduction helpers
// ---------------------------------------------------------------------------
__global__ void k_cvt_bf16(const float* __restrict__ s,
                           unsigned short* __restrict__ d, int n) {
  int i = blockIdx.x * blockDim.x + threadIdx.x;
  if (i < n) d[i] = f2bf(s[i]);
}

// RMSNorm: fp32 in, bf16 out (all consumers are WMMA ops)
__global__ __launch_bounds__(256) void k_rms(const float* __restrict__ x,
    const float* __restrict__ g, unsigned short* __restrict__ out, int D) {
  __shared__ float red[256];
  int row = blockIdx.x;
  const float* xr = x + (size_t)row * D;
  float s = 0.f;
  for (int i = threadIdx.x; i < D; i += 256) { float v = xr[i]; s += v * v; }
  red[threadIdx.x] = s; __syncthreads();
  for (int st = 128; st > 0; st >>= 1) {
    if (threadIdx.x < st) red[threadIdx.x] += red[threadIdx.x + st];
    __syncthreads();
  }
  float inv = rsqrtf(red[0] / (float)D + 1e-6f);
  unsigned short* orow = out + (size_t)row * D;
  for (int i = threadIdx.x; i < D; i += 256) orow[i] = f2bf(xr[i] * g[i] * inv);
}

__global__ void k_gather_rows_f32(const float* __restrict__ src,
    const int* __restrict__ idx, float* __restrict__ dst, int D) {
  int row = blockIdx.x;
  const float* s = src + (size_t)idx[row] * D;
  float* d = dst + (size_t)row * D;
  for (int i = threadIdx.x; i < D; i += blockDim.x) d[i] = s[i];
}

__global__ void k_gather_rows_bf16(const unsigned short* __restrict__ src,
    const int* __restrict__ idx, unsigned short* __restrict__ dst, int D) {
  int row = blockIdx.x;
  const uint32_t* s = (const uint32_t*)(src + (size_t)idx[row] * D);
  uint32_t* d = (uint32_t*)(dst + (size_t)row * D);
  for (int i = threadIdx.x; i < D / 2; i += blockDim.x) d[i] = s[i];
}

__global__ void k_add_inplace(float* __restrict__ x, const float* __restrict__ y, int n) {
  int i = blockIdx.x * blockDim.x + threadIdx.x;
  if (i < n) x[i] += y[i];
}

__global__ void k_add_out(float* __restrict__ d, const float* __restrict__ a,
                          const float* __restrict__ b, int n) {
  int i = blockIdx.x * blockDim.x + threadIdx.x;
  if (i < n) d[i] = a[i] + b[i];
}

__global__ void k_gelu_bf16(unsigned short* __restrict__ x, int n) {
  int i = blockIdx.x * blockDim.x + threadIdx.x;
  if (i < n) {
    float v = bf2f(x[i]);
    float c = 0.7978845608028654f * (v + 0.044715f * v * v * v);
    x[i] = f2bf(0.5f * v * (1.0f + tanhf(c)));
  }
}

__global__ void k_zero(float* p) { p[0] = 0.0f; }

__global__ void k_concat_i32(const int* __restrict__ a, int na,
                             const int* __restrict__ b, int nb, int* __restrict__ out) {
  int i = blockIdx.x * blockDim.x + threadIdx.x;
  if (i < na) out[i] = a[i];
  else if (i < na + nb) out[i] = b[i - na];
}

__device__ __forceinline__ float block_reduce(float v, float* red, int domax) {
  red[threadIdx.x] = v; __syncthreads();
  for (int st = 128; st > 0; st >>= 1) {
    if (threadIdx.x < st)
      red[threadIdx.x] = domax ? fmaxf(red[threadIdx.x], red[threadIdx.x + st])
                               : (red[threadIdx.x] + red[threadIdx.x + st]);
    __syncthreads();
  }
  float r = red[0]; __syncthreads();
  return r;
}

// KD loss: forward KL(teacher || student) per row, masked, atomically summed.
__global__ __launch_bounds__(256) void k_kd(const float* __restrict__ lt,
    const float* __restrict__ ls, const int* __restrict__ labels,
    const int* __restrict__ nitems, float* __restrict__ out, int V) {
  int row = blockIdx.x;
  if (labels[row] == -100) return;
  __shared__ float red[256];
  const float* t = lt + (size_t)row * V;
  const float* s = ls + (size_t)row * V;
  float mt = -3.0e38f, ms = -3.0e38f;
  for (int i = threadIdx.x; i < V; i += 256) {
    mt = fmaxf(mt, t[i]);
    ms = fmaxf(ms, s[i]);
  }
  mt = block_reduce(mt, red, 1);
  ms = block_reduce(ms, red, 1);
  float zt = 0.f, zs = 0.f;
  for (int i = threadIdx.x; i < V; i += 256) {
    zt += __expf(t[i] - mt);
    zs += __expf(s[i] - ms);
  }
  zt = block_reduce(zt, red, 0);
  zs = block_reduce(zs, red, 0);
  float lzt = logf(zt), lzs = logf(zs);
  float kl = 0.f;
  for (int i = threadIdx.x; i < V; i += 256) {
    float lpt = t[i] - mt - lzt;
    float lps = s[i] - ms - lzs;
    kl += __expf(lpt) * (lpt - lps);
  }
  kl = block_reduce(kl, red, 0);
  if (threadIdx.x == 0) atomicAdd(out, kl / (float)nitems[0]);
}

// ---------------------------------------------------------------------------
// Host orchestration
// ---------------------------------------------------------------------------
extern "C" void kernel_launch(void* const* d_in, const int* in_sizes, int n_in,
                              void* d_out, int out_size, void* d_ws, size_t ws_size,
                              hipStream_t stream) {
  (void)in_sizes; (void)n_in; (void)out_size; (void)ws_size;
  const int P = P_TOK, T = T_TOK, D = DIM, V = VOC, FF = FFDIM, L = NLAY, H = HEADS;
  const int D3 = 3 * D;

  const int* prefix_input_ids    = (const int*)d_in[0];
  const int* prefix_batch_ids    = (const int*)d_in[1];
  const int* prefix_position_ids = (const int*)d_in[2];
  const int* input_ids           = (const int*)d_in[3];
  const int* batch_ids           = (const int*)d_in[4];
  const int* position_ids        = (const int*)d_in[5];
  const int* tail_gather         = (const int*)d_in[6];
  const int* labels              = (const int*)d_in[7];
  const int* nitems              = (const int*)d_in[8];
  const float* Wt_embed = (const float*)d_in[9];
  const float* Wt_qkv   = (const float*)d_in[10];
  const float* Wt_o     = (const float*)d_in[11];
  const float* Wt_m1    = (const float*)d_in[12];
  const float* Wt_m2    = (const float*)d_in[13];
  const float* gt_ln1   = (const float*)d_in[14];
  const float* gt_ln2   = (const float*)d_in[15];
  const float* gt_lnf   = (const float*)d_in[16];
  const float* Wd_embed = (const float*)d_in[17];
  const float* Wd_qkv   = (const float*)d_in[18];
  const float* Wd_o     = (const float*)d_in[19];
  const float* Wd_m1    = (const float*)d_in[20];
  const float* Wd_m2    = (const float*)d_in[21];
  const float* gd_ln1   = (const float*)d_in[22];
  const float* gd_ln2   = (const float*)d_in[23];
  const float* gd_lnf   = (const float*)d_in[24];
  float* out = (float*)d_out;

  // Deterministic bump allocator on the workspace
  char* wp = (char*)d_ws;
  auto alloc = [&](size_t bytes) -> void* {
    void* p = (void*)wp;
    wp += (bytes + 255) & ~(size_t)255;
    return p;
  };
  typedef unsigned short u16;
  // fp32 activations (residual streams + logits)
  float* x    = (float*)alloc((size_t)P * D * 4);
  float* tmp  = (float*)alloc((size_t)P * D * 4);
  float* xq   = (float*)alloc((size_t)T * D * 4);
  float* y    = (float*)alloc((size_t)T * D * 4);
  float* lt   = (float*)alloc((size_t)T * V * 4);
  float* ls   = (float*)alloc((size_t)T * V * 4);
  // bf16 activations
  u16* xn   = (u16*)alloc((size_t)(P + T) * D * 2);   // norm out / draft kv-norm
  u16* qkv  = (u16*)alloc((size_t)(P + T) * D3 * 2);
  u16* aout = (u16*)alloc((size_t)P * D * 2);
  u16* mlp  = (u16*)alloc((size_t)P * FF * 2);
  u16* rmsf = (u16*)alloc((size_t)P * D * 2);
  u16* xg   = (u16*)alloc((size_t)T * D * 2);
  // bf16 weights (converted once per call; halves HBM traffic, vocab B fits L2)
  u16* bWt_embed = (u16*)alloc((size_t)V * D * 2);
  u16* bWt_qkv   = (u16*)alloc((size_t)L * D * D3 * 2);
  u16* bWt_o     = (u16*)alloc((size_t)L * D * D * 2);
  u16* bWt_m1    = (u16*)alloc((size_t)L * D * FF * 2);
  u16* bWt_m2    = (u16*)alloc((size_t)L * FF * D * 2);
  u16* bWd_embed = (u16*)alloc((size_t)V * D * 2);
  u16* bWd_qkv   = (u16*)alloc((size_t)D * D3 * 2);
  u16* bWd_o     = (u16*)alloc((size_t)D * D * 2);
  u16* bWd_m1    = (u16*)alloc((size_t)D * FF * 2);
  u16* bWd_m2    = (u16*)alloc((size_t)FF * D * 2);
  int* full_b = (int*)alloc((size_t)(P + T) * 4);
  int* full_p = (int*)alloc((size_t)(P + T) * 4);

  auto cvt = [&](const float* s, u16* d, size_t n) {
    k_cvt_bf16<<<(unsigned)((n + 255) / 256), 256, 0, stream>>>(s, d, (int)n);
  };
  cvt(Wt_embed, bWt_embed, (size_t)V * D);
  cvt(Wt_qkv, bWt_qkv, (size_t)L * D * D3);
  cvt(Wt_o, bWt_o, (size_t)L * D * D);
  cvt(Wt_m1, bWt_m1, (size_t)L * D * FF);
  cvt(Wt_m2, bWt_m2, (size_t)L * FF * D);
  cvt(Wd_embed, bWd_embed, (size_t)V * D);
  cvt(Wd_qkv, bWd_qkv, (size_t)D * D3);
  cvt(Wd_o, bWd_o, (size_t)D * D);
  cvt(Wd_m1, bWd_m1, (size_t)D * FF);
  cvt(Wd_m2, bWd_m2, (size_t)FF * D);

  const float scale = 1.0f / 16.0f;  // 1/sqrt(DH)

  auto gemm = [&](const u16* A, const u16* B, void* C, int M, int N, int K,
                  int btrans, int cbf16) {
    dim3 g(N / GBN, M / GBM);
    k_gemm_bf16<<<g, 256, 0, stream>>>(A, B, C, M, N, K, btrans, cbf16);
  };

  // -------- target causal prefill (L layers) --------
  k_gather_rows_f32<<<P, 256, 0, stream>>>(Wt_embed, prefix_input_ids, x, D);
  for (int l = 0; l < L; ++l) {
    k_rms<<<P, 256, 0, stream>>>(x, gt_ln1 + (size_t)l * D, xn, D);
    gemm(xn, bWt_qkv + (size_t)l * D * D3, qkv, P, D3, D, 0, 1);
    dim3 fg(P / FQ, H);
    k_flash<<<fg, 128, 0, stream>>>(qkv, D3, qkv + D, D3, qkv + 2 * D, D3,
        aout, D, P, P, prefix_batch_ids, prefix_position_ids,
        prefix_batch_ids, prefix_position_ids, 0, P, scale);
    gemm(aout, bWt_o + (size_t)l * D * D, tmp, P, D, D, 0, 0);
    int n = P * D;
    k_add_inplace<<<(n + 255) / 256, 256, 0, stream>>>(x, tmp, n);
    k_rms<<<P, 256, 0, stream>>>(x, gt_ln2 + (size_t)l * D, xn, D);
    gemm(xn, bWt_m1 + (size_t)l * D * FF, mlp, P, FF, D, 0, 1);
    int nf = P * FF;
    k_gelu_bf16<<<(nf + 255) / 256, 256, 0, stream>>>(mlp, nf);
    gemm(mlp, bWt_m2 + (size_t)l * FF * D, tmp, P, D, FF, 0, 0);
    k_add_inplace<<<(n + 255) / 256, 256, 0, stream>>>(x, tmp, n);
  }
  // teacher logits: gather the 1024 needed rows BEFORE the vocab projection
  k_rms<<<P, 256, 0, stream>>>(x, gt_lnf, rmsf, D);
  k_gather_rows_bf16<<<T, 256, 0, stream>>>(rmsf, tail_gather, xg, D);
  gemm(xg, bWt_embed, lt, T, V, D, 1, 0);  // @ Wt_embed^T

  // -------- draft forward --------
  k_gather_rows_f32<<<T, 256, 0, stream>>>(Wd_embed, input_ids, xq, D);
  // rms(xkv, gd_ln1) with xkv = [hidden ; xq]; draft q rows == kv rows P..P+T
  k_rms<<<P, 256, 0, stream>>>(x, gd_ln1, xn, D);
  k_rms<<<T, 256, 0, stream>>>(xq, gd_ln1, xn + (size_t)P * D, D);
  gemm(xn, bWd_qkv, qkv, P + T, D3, D, 0, 1);
  {
    int n = P + T;
    k_concat_i32<<<(n + 255) / 256, 256, 0, stream>>>(prefix_batch_ids, P, batch_ids, T, full_b);
    k_concat_i32<<<(n + 255) / 256, 256, 0, stream>>>(prefix_position_ids, P, position_ids, T, full_p);
  }
  dim3 fg(T / FQ, H);
  k_flash<<<fg, 128, 0, stream>>>(qkv + (size_t)P * D3, D3, qkv + D, D3, qkv + 2 * D, D3,
      aout, D, T, P + T, batch_ids, position_ids, full_b, full_p, 1, P, scale);
  gemm(aout, bWd_o, tmp, T, D, D, 0, 0);
  int nT = T * D;
  k_add_out<<<(nT + 255) / 256, 256, 0, stream>>>(y, xq, tmp, nT);
  k_rms<<<T, 256, 0, stream>>>(y, gd_ln2, xn, D);
  gemm(xn, bWd_m1, mlp, T, FF, D, 0, 1);
  int nf = T * FF;
  k_gelu_bf16<<<(nf + 255) / 256, 256, 0, stream>>>(mlp, nf);
  gemm(mlp, bWd_m2, tmp, T, D, FF, 0, 0);
  k_add_inplace<<<(nT + 255) / 256, 256, 0, stream>>>(y, tmp, nT);
  k_rms<<<T, 256, 0, stream>>>(y, gd_lnf, xn, D);
  gemm(xn, bWd_embed, ls, T, V, D, 1, 0);  // @ Wd_embed^T

  // -------- KD loss --------
  k_zero<<<1, 1, 0, stream>>>(out);
  k_kd<<<T, 256, 0, stream>>>(lt, ls, labels, nitems, out, V);
}
